// IPA_45956150067208
// MI455X (gfx1250) — compile-verified, matrix operand order checked by reading, not run on hardware
//
#include <hip/hip_runtime.h>

#define N_    1024
#define NODE_ 384
#define PAIR_ 128
#define H_    12
#define QKVO_ 1152   // 2*H*DQS + H*DVS + 2*H*DQP*3 + H*DVP*3
#define OUTW_ 2112   // H*(PAIR + DVS + 4*DVP)

#define SCALAR_SCALE 0.14433756729740643f   // (3*16)^-0.5
#define POINT_SCALE  0.08091518860166042f   // (3*4*9*sqrt(2))^-0.5
#define PAIR_SCALE   0.5773502691896258f    // 3^-0.5
#define LN_EPS  1e-5f
#define NORM_EPS 1e-8f
#define MASK_MIN -1.0e30f

typedef __bf16 bf16_t;
typedef __attribute__((ext_vector_type(16))) __bf16 v16bf;
typedef __attribute__((ext_vector_type(8)))  __bf16 v8bf;
typedef __attribute__((ext_vector_type(8)))  float  v8f;
typedef __attribute__((ext_vector_type(4)))  unsigned int v4u;
typedef __attribute__((ext_vector_type(8)))  int v8i;
typedef __attribute__((ext_vector_type(4)))  int v4i;

#if __has_builtin(__builtin_amdgcn_tensor_load_to_lds) && \
    __has_builtin(__builtin_amdgcn_s_wait_tensorcnt)
#define USE_TDM 1
#else
#define USE_TDM 0
#endif

static __device__ inline v8f vzero8() {
  v8f z;
#pragma unroll
  for (int t = 0; t < 8; ++t) z[t] = 0.0f;
  return z;
}
static __device__ inline v16bf bzero16() {
  v16bf z;
#pragma unroll
  for (int t = 0; t < 16; ++t) z[t] = (bf16_t)0.0f;
  return z;
}
static __device__ inline v8f wmma_bf16(v16bf a, v16bf b, v8f c) {
  return __builtin_amdgcn_wmma_f32_16x16x32_bf16(false, a, false, b, (short)0, c,
                                                 false, false);
}
// Fragment from a contiguous 32-element K-slice of a row-major row.
// CDNA5 16-bit A/B layout: lanes 0-15 hold K {0..7,16..23}, lanes 16-31 hold
// K {8..15,24..31}.  Two 16-byte vector loads per lane.
static __device__ inline v16bf frag_row(const bf16_t* rowk, int lane) {
  int kb = (lane >> 4) << 3;
  v8bf lo = *(const v8bf*)(rowk + kb);
  v8bf hi = *(const v8bf*)(rowk + 16 + kb);
  v16bf r;
#pragma unroll
  for (int t = 0; t < 8; ++t) { r[t] = lo[t]; r[t + 8] = hi[t]; }
  return r;
}

#if USE_TDM
// TDM descriptor: 2-D tile, 32 rows x 128 f32, row stride 128, into LDS.
// 6-arg builtin form: (v4u g0, v8i g1, v4i g2, v4i g3, v8i g4, i32 cpol)
static __device__ inline void tdm_load_pair_tile(const float* gsrc, unsigned lds_byte) {
  unsigned long long ga = (unsigned long long)(size_t)gsrc;
  v4u g0;
  g0[0] = 1u;                                  // count=1 (valid user descriptor)
  g0[1] = lds_byte;                            // lds_addr
  g0[2] = (unsigned)ga;                        // global_addr[31:0]
  g0[3] = (unsigned)((ga >> 32) & 0x01FFFFFFu) | (2u << 30);  // addr[56:32] | type=2
  v8i g1;
  g1[0] = (int)(2u << 16);                     // data_size = 4 bytes
  g1[1] = (int)(128u << 16);                   // tensor_dim0 = 128 (bits 79:48)
  g1[2] = (int)(32u << 16);                    // tensor_dim1 = 32  (bits 111:80)
  g1[3] = (int)(128u << 16);                   // tile_dim0 = 128   (bits 127:112)
  g1[4] = 32;                                  // tile_dim1 = 32
  g1[5] = 128;                                 // tensor_dim0_stride = 128
  g1[6] = 0;
  g1[7] = 0;
  v4i z4; z4[0] = z4[1] = z4[2] = z4[3] = 0;
  v8i z8;
#pragma unroll
  for (int t = 0; t < 8; ++t) z8[t] = 0;
  __builtin_amdgcn_tensor_load_to_lds(g0, g1, z4, z4, z8, 0);
}
#endif

// ---------------- prep kernels ----------------

__global__ void k_ln_node(const float* __restrict__ x, const float* __restrict__ g,
                          const float* __restrict__ bb, bf16_t* __restrict__ nf) {
  __shared__ float r1[128], r2[128];
  int n = blockIdx.x, t = threadIdx.x;
  const float* row = x + (size_t)n * NODE_;
  float v0 = row[t], v1 = row[t + 128], v2 = row[t + 256];
  r1[t] = v0 + v1 + v2;
  r2[t] = v0 * v0 + v1 * v1 + v2 * v2;
  __syncthreads();
  for (int off = 64; off > 0; off >>= 1) {
    if (t < off) { r1[t] += r1[t + off]; r2[t] += r2[t + off]; }
    __syncthreads();
  }
  float mean = r1[0] / (float)NODE_;
  float var  = r2[0] / (float)NODE_ - mean * mean;
  float rstd = rsqrtf(var + LN_EPS);
  bf16_t* orow = nf + (size_t)n * NODE_;
  orow[t]       = (bf16_t)((v0 - mean) * rstd * g[t]       + bb[t]);
  orow[t + 128] = (bf16_t)((v1 - mean) * rstd * g[t + 128] + bb[t + 128]);
  orow[t + 256] = (bf16_t)((v2 - mean) * rstd * g[t + 256] + bb[t + 256]);
}

__global__ void k_f2b(const float* __restrict__ src, bf16_t* __restrict__ dst, int count) {
  int i = blockIdx.x * blockDim.x + threadIdx.x;
  if (i < count) dst[i] = (bf16_t)src[i];
}

// W_pb (12x128) -> padded (16x128) bf16, pre-scaled by PAIR_SCALE
__global__ void k_wpb(const float* __restrict__ src, bf16_t* __restrict__ dst) {
  int i = blockIdx.x * blockDim.x + threadIdx.x;  // 2048 total
  int h = i >> 7, c = i & 127;
  dst[i] = (h < H_) ? (bf16_t)(src[h * PAIR_ + c] * PAIR_SCALE) : (bf16_t)0.0f;
}

// ---------------- qkv = nf @ W_qkv^T  (WMMA) ----------------

__global__ void k_gemm_qkv(const bf16_t* __restrict__ nf, const bf16_t* __restrict__ wq,
                           float* __restrict__ qkv) {
  int wg = blockIdx.x * 4 + (threadIdx.x >> 5);  // 4608 waves = 64 x 72 tiles
  int lane = threadIdx.x & 31;
  int tm = wg & 63, tn = wg >> 6;
  const bf16_t* arow = nf + (size_t)(tm * 16 + (lane & 15)) * NODE_;
  const bf16_t* brow = wq + (size_t)(tn * 16 + (lane & 15)) * NODE_;
  v8f acc = vzero8();
  for (int k0 = 0; k0 < NODE_; k0 += 32)
    acc = wmma_bf16(frag_row(arow + k0, lane), frag_row(brow + k0, lane), acc);
  int o = tn * 16 + (lane & 15);
  int rb = tm * 16 + ((lane >> 4) << 3);
#pragma unroll
  for (int v = 0; v < 8; ++v) qkv[(size_t)(rb + v) * QKVO_ + o] = acc[v];
}

// ---------------- rigid transforms + operand packing ----------------
// qk_a[h][n][32]  = [SCALAR_SCALE*q_s | 2*POINT_SCALE*gamma*qp_glob | 0 pad]
// qk_b[h][n][32]  = [k_s | kp_glob | 0 pad]
// vsvpT[h][dim][n] (dim<48) = [v_s(16) | vp_glob(24) | 0(8)]  -- K-major for WMMA B
__global__ void k_points(const float* __restrict__ qkv, const float* __restrict__ rot,
                         const float* __restrict__ trans, const float* __restrict__ pw,
                         bf16_t* __restrict__ qk_a, bf16_t* __restrict__ qk_b,
                         bf16_t* __restrict__ vsvpT, float* __restrict__ qrow,
                         float* __restrict__ kcol) {
  int n = blockIdx.x;
  int h = threadIdx.x >> 5, lane = threadIdx.x & 31;
  const float* row = qkv + (size_t)n * QKVO_;
  float R[9];
#pragma unroll
  for (int t = 0; t < 9; ++t) R[t] = rot[(size_t)n * 9 + t];
  float T0 = trans[(size_t)n * 3 + 0], T1 = trans[(size_t)n * 3 + 1],
        T2 = trans[(size_t)n * 3 + 2];
  float gamma = log1pf(expf(pw[h]));  // softplus(point_weights)
  size_t bhn = (size_t)h * N_ + n;
  if (lane < 16) {
    float qs = row[h * 16 + lane];
    float ks = row[192 + h * 16 + lane];
    float vs = row[384 + h * 16 + lane];
    qk_a[bhn * 32 + lane] = (bf16_t)(qs * SCALAR_SCALE);
    qk_b[bhn * 32 + lane] = (bf16_t)ks;
    vsvpT[((size_t)h * 48 + lane) * N_ + n] = (bf16_t)vs;
  }
  int d = lane / 3, r = lane - 3 * d;
  if (lane < 12) {
    const float* qp = row + 576 + (h * 4 + d) * 3;
    const float* kp = row + 720 + (h * 4 + d) * 3;
    float Tr = (r == 0) ? T0 : ((r == 1) ? T1 : T2);
    float qg = R[r * 3] * qp[0] + R[r * 3 + 1] * qp[1] + R[r * 3 + 2] * qp[2] + Tr;
    float kg = R[r * 3] * kp[0] + R[r * 3 + 1] * kp[1] + R[r * 3 + 2] * kp[2] + Tr;
    qk_a[bhn * 32 + 16 + lane] = (bf16_t)(qg * (2.0f * POINT_SCALE * gamma));
    qk_b[bhn * 32 + 16 + lane] = (bf16_t)kg;
  } else if (lane < 16) {
    qk_a[bhn * 32 + 16 + lane] = (bf16_t)0.0f;
    qk_b[bhn * 32 + 16 + lane] = (bf16_t)0.0f;
  }
  if (lane < 24) {
    const float* vp = row + 864 + (h * 8 + d) * 3;
    float Tr = (r == 0) ? T0 : ((r == 1) ? T1 : T2);
    float vg = R[r * 3] * vp[0] + R[r * 3 + 1] * vp[1] + R[r * 3 + 2] * vp[2] + Tr;
    vsvpT[((size_t)h * 48 + 16 + lane) * N_ + n] = (bf16_t)vg;
  } else {
    vsvpT[((size_t)h * 48 + 16 + lane) * N_ + n] = (bf16_t)0.0f;
  }
  if (lane == 0) {
    float q2 = 0.0f, k2 = 0.0f;
#pragma unroll
    for (int dd = 0; dd < 4; ++dd) {
      const float* qp = row + 576 + (h * 4 + dd) * 3;
      const float* kp = row + 720 + (h * 4 + dd) * 3;
#pragma unroll
      for (int rr = 0; rr < 3; ++rr) {
        float Tr = (rr == 0) ? T0 : ((rr == 1) ? T1 : T2);
        float qg = R[rr * 3] * qp[0] + R[rr * 3 + 1] * qp[1] + R[rr * 3 + 2] * qp[2] + Tr;
        float kg = R[rr * 3] * kp[0] + R[rr * 3 + 1] * kp[1] + R[rr * 3 + 2] * kp[2] + Tr;
        q2 += qg * qg; k2 += kg * kg;
      }
    }
    qrow[bhn] = -POINT_SCALE * gamma * q2;
    kcol[bhn] = -POINT_SCALE * gamma * k2;
  }
}

// ---------------- pair bias: layernorm(pf) @ W_pb^T  (WMMA) ----------------
__global__ void k_pairbias(const float* __restrict__ pair, const float* __restrict__ g,
                           const float* __restrict__ bb, const bf16_t* __restrict__ wpb,
                           float* __restrict__ bias /* [12][N][N], pre-scaled */) {
  __shared__ bf16_t tile[2][16][PAIR_];
  int i = blockIdx.x;
  int w = threadIdx.x >> 5, lane = threadIdx.x & 31;
  int jt = blockIdx.y * 2 + w;
  int r = lane & 15, hf = lane >> 4;
  const float* row = pair + ((size_t)i * N_ + jt * 16 + r) * PAIR_;
  float s = 0.0f, s2 = 0.0f;
  for (int c = hf * 64; c < hf * 64 + 64; ++c) { float x = row[c]; s += x; s2 += x * x; }
  s += __shfl_xor(s, 16, 32);
  s2 += __shfl_xor(s2, 16, 32);
  float mean = s / (float)PAIR_;
  float rstd = rsqrtf(s2 / (float)PAIR_ - mean * mean + LN_EPS);
  for (int c = hf * 64; c < hf * 64 + 64; ++c)
    tile[w][r][c] = (bf16_t)((row[c] - mean) * rstd * g[c] + bb[c]);
  __syncthreads();
  v8f acc = vzero8();
  int kb = hf << 3;
#pragma unroll
  for (int sstep = 0; sstep < 4; ++sstep) {
    v16bf a = frag_row(wpb + (size_t)r * PAIR_ + sstep * 32, lane);  // A row = head
    v16bf b;
#pragma unroll
    for (int t = 0; t < 8; ++t) {
      b[t]     = tile[w][r][sstep * 32 + kb + t];
      b[t + 8] = tile[w][r][sstep * 32 + 16 + kb + t];
    }
    acc = wmma_bf16(a, b, acc);
  }
  int j = jt * 16 + r;
#pragma unroll
  for (int v = 0; v < 8; ++v) {
    int hh = (hf << 3) + v;
    if (hh < H_) bias[((size_t)hh * N_ + i) * N_ + j] = acc[v];
  }
}

// ---------------- fused logits (WMMA): scalar QK + point QK + biases ------
// one wave per (h, i-tile, 4 j-tiles); A fragment + row biases reused 4x.
__global__ void k_logits(const bf16_t* __restrict__ qk_a, const bf16_t* __restrict__ qk_b,
                         const float* __restrict__ qrow, const float* __restrict__ kcol,
                         const int* __restrict__ mask, float* __restrict__ logits) {
  int it = blockIdx.y, h = blockIdx.z;
  int lane = threadIdx.x;
  v16bf a = frag_row(qk_a + ((size_t)h * N_ + it * 16 + (lane & 15)) * 32, lane);
  int rb = it * 16 + ((lane >> 4) << 3);
  float qr[8];
  int mi[8];
#pragma unroll
  for (int v = 0; v < 8; ++v) {
    qr[v] = qrow[(size_t)h * N_ + rb + v];
    mi[v] = mask[rb + v];
  }
#pragma unroll
  for (int jj = 0; jj < 4; ++jj) {
    int jt = blockIdx.x * 4 + jj;
    v16bf b = frag_row(qk_b + ((size_t)h * N_ + jt * 16 + (lane & 15)) * 32, lane);
    v8f d = wmma_bf16(a, b, vzero8());
    int j = jt * 16 + (lane & 15);
    float kc = kcol[(size_t)h * N_ + j];
    int mj = mask[j];
#pragma unroll
    for (int v = 0; v < 8; ++v) {
      size_t idx = ((size_t)h * N_ + rb + v) * N_ + j;
      float val = d[v] + qr[v] + kc + logits[idx];
      if (!(mi[v] && mj)) val = MASK_MIN;
      logits[idx] = val;
    }
  }
}

// ---------------- softmax rows -> attn bf16 ----------------
__global__ void k_softmax(const float* __restrict__ logits, bf16_t* __restrict__ attn) {
  __shared__ float red[256];
  int i = blockIdx.x, h = blockIdx.y, t = threadIdx.x;
  const float* row = logits + ((size_t)h * N_ + i) * N_;
  float m = -3.4e38f;
  for (int j = t; j < N_; j += 256) m = fmaxf(m, row[j]);
  red[t] = m; __syncthreads();
  for (int off = 128; off > 0; off >>= 1) {
    if (t < off) red[t] = fmaxf(red[t], red[t + off]);
    __syncthreads();
  }
  float mx = red[0]; __syncthreads();
  float s = 0.0f;
  for (int j = t; j < N_; j += 256) s += expf(row[j] - mx);
  red[t] = s; __syncthreads();
  for (int off = 128; off > 0; off >>= 1) {
    if (t < off) red[t] += red[t + off];
    __syncthreads();
  }
  float inv = 1.0f / red[0];
  bf16_t* orow = attn + ((size_t)h * N_ + i) * N_;
  for (int j = t; j < N_; j += 256) orow[j] = (bf16_t)(expf(row[j] - mx) * inv);
}

// ---------------- attn @ [v_s | vp]  (WMMA, contiguous B frags) -----------
__global__ void k_av(const bf16_t* __restrict__ attn, const bf16_t* __restrict__ vsvpT,
                     bf16_t* __restrict__ outb, float* __restrict__ resptg) {
  int it = blockIdx.x, h = blockIdx.y;
  int w = threadIdx.x >> 5, lane = threadIdx.x & 31;
  int colf = w * 16 + (lane & 15);  // 0..47
  v8f acc = vzero8();
  const bf16_t* arow = attn + ((size_t)h * N_ + it * 16 + (lane & 15)) * N_;
  const bf16_t* brow = vsvpT + ((size_t)h * 48 + colf) * N_;
  for (int k0 = 0; k0 < N_; k0 += 32)
    acc = wmma_bf16(frag_row(arow + k0, lane), frag_row(brow + k0, lane), acc);
  int rb = it * 16 + ((lane >> 4) << 3);
#pragma unroll
  for (int v = 0; v < 8; ++v) {
    int i = rb + v;
    if (w == 0) {
      outb[(size_t)i * OUTW_ + h * 16 + (lane & 15)] = (bf16_t)acc[v];
    } else {
      int dim = (w - 1) * 16 + (lane & 15);
      if (dim < 24) resptg[((size_t)i * H_ + h) * 24 + dim] = acc[v];
    }
  }
}

// ---------------- inverse rigid + point norms ----------------
__global__ void k_inv(const float* __restrict__ resptg, const float* __restrict__ rot,
                      const float* __restrict__ trans, bf16_t* __restrict__ outb) {
  int n = blockIdx.x;
  int h = threadIdx.x >> 3, d = threadIdx.x & 7;  // 96 threads
  const float* R = rot + (size_t)n * 9;
  const float* p = resptg + ((size_t)n * H_ + h) * 24 + d * 3;
  float x0 = p[0] - trans[(size_t)n * 3 + 0];
  float x1 = p[1] - trans[(size_t)n * 3 + 1];
  float x2 = p[2] - trans[(size_t)n * 3 + 2];
  float l0 = R[0] * x0 + R[3] * x1 + R[6] * x2;  // R^T (x - t)
  float l1 = R[1] * x0 + R[4] * x1 + R[7] * x2;
  float l2 = R[2] * x0 + R[5] * x1 + R[8] * x2;
  bf16_t* o = outb + (size_t)n * OUTW_;
  o[192 + h * 24 + d * 3 + 0] = (bf16_t)l0;
  o[192 + h * 24 + d * 3 + 1] = (bf16_t)l1;
  o[192 + h * 24 + d * 3 + 2] = (bf16_t)l2;
  o[480 + h * 8 + d] = (bf16_t)sqrtf(l0 * l0 + l1 * l1 + l2 * l2 + NORM_EPS);
}

// ---------------- res_pair = attn @ layernorm(pf_i)  (WMMA + TDM) ----------
// one block (8 waves) per query i; A rows = heads (pad 12->16), K = j (1024).
// pair_feats tiles (32x128 f32) are DMA'd into LDS by the Tensor Data Mover,
// double-buffered against the normalize+WMMA of the previous tile.
__global__ void k_respair(const float* __restrict__ pair, const float* __restrict__ g,
                          const float* __restrict__ bb, const bf16_t* __restrict__ attn,
                          bf16_t* __restrict__ outb) {
#if USE_TDM
  __shared__ float raw[2][32][PAIR_];
#endif
  __shared__ bf16_t tile[32][PAIR_];
  __shared__ float part[32][8][2];
  __shared__ float stat[32][2];
  int i = blockIdx.x;
  int tid = threadIdx.x;
  int w = tid >> 5, lane = tid & 31;
  int h = lane & 15;
  int colf = w * 16 + (lane & 15);
  int rr = tid >> 3, sub = tid & 7;  // 32 rows x 8 threads, 16 elems each
  v8f acc = vzero8();
  int kb = (lane >> 4) << 3;
  const float* src = pair + (size_t)i * N_ * PAIR_;
#if USE_TDM
  if (tid < 32) tdm_load_pair_tile(src, (unsigned)(size_t)&raw[0][0][0]);
#endif
  for (int kt = 0; kt < 32; ++kt) {
#if USE_TDM
    if (tid < 32) {
      if (kt + 1 < 32) {
        tdm_load_pair_tile(src + (size_t)(kt + 1) * 32 * PAIR_,
                           (unsigned)(size_t)&raw[(kt + 1) & 1][0][0]);
        __builtin_amdgcn_s_wait_tensorcnt(1);
      } else {
        __builtin_amdgcn_s_wait_tensorcnt(0);
      }
    }
    __syncthreads();
    const float* row = &raw[kt & 1][rr][sub * 16];
#else
    const float* row = src + ((size_t)kt * 32 + rr) * PAIR_ + sub * 16;
#endif
    float x[16], s = 0.0f, s2 = 0.0f;
#pragma unroll
    for (int t = 0; t < 16; ++t) { x[t] = row[t]; s += x[t]; s2 += x[t] * x[t]; }
    part[rr][sub][0] = s; part[rr][sub][1] = s2;
    __syncthreads();
    if (sub == 0) {
      float ts = 0.0f, ts2 = 0.0f;
#pragma unroll
      for (int t = 0; t < 8; ++t) { ts += part[rr][t][0]; ts2 += part[rr][t][1]; }
      float mean = ts / (float)PAIR_;
      stat[rr][0] = mean;
      stat[rr][1] = rsqrtf(ts2 / (float)PAIR_ - mean * mean + LN_EPS);
    }
    __syncthreads();
    float mean = stat[rr][0], rstd = stat[rr][1];
#pragma unroll
    for (int t = 0; t < 16; ++t)
      tile[rr][sub * 16 + t] = (bf16_t)((x[t] - mean) * rstd * g[sub * 16 + t] + bb[sub * 16 + t]);
    __syncthreads();
    v16bf a = (h < H_) ? frag_row(attn + ((size_t)h * N_ + i) * N_ + kt * 32, lane)
                       : bzero16();
    v16bf b;
#pragma unroll
    for (int t = 0; t < 8; ++t) {
      b[t]     = tile[kb + t][colf];
      b[t + 8] = tile[16 + kb + t][colf];
    }
    acc = wmma_bf16(a, b, acc);
    __syncthreads();
  }
#pragma unroll
  for (int v = 0; v < 8; ++v) {
    int hh = ((lane >> 4) << 3) + v;
    if (hh < H_) outb[(size_t)i * OUTW_ + 576 + hh * PAIR_ + colf] = (bf16_t)acc[v];
  }
}

// ---------------- final GEMM: out = concat @ W_out^T + b_out (WMMA) -------
__global__ void k_outgemm(const bf16_t* __restrict__ outb, const bf16_t* __restrict__ wout,
                          const float* __restrict__ bout, float* __restrict__ out) {
  int wg = blockIdx.x * 4 + (threadIdx.x >> 5);  // 1536 waves = 64 x 24 tiles
  int lane = threadIdx.x & 31;
  int tm = wg & 63, tn = wg >> 6;
  const bf16_t* arow = outb + (size_t)(tm * 16 + (lane & 15)) * OUTW_;
  const bf16_t* brow = wout + (size_t)(tn * 16 + (lane & 15)) * OUTW_;
  v8f acc = vzero8();
  for (int k0 = 0; k0 < OUTW_; k0 += 32)
    acc = wmma_bf16(frag_row(arow + k0, lane), frag_row(brow + k0, lane), acc);
  int o = tn * 16 + (lane & 15);
  float bo = bout[o];
  int rb = tm * 16 + ((lane >> 4) << 3);
#pragma unroll
  for (int v = 0; v < 8; ++v) out[(size_t)(rb + v) * NODE_ + o] = acc[v] + bo;
}

// ---------------- host ----------------

static inline size_t align256(size_t x) { return (x + 255) & ~(size_t)255; }

extern "C" void kernel_launch(void* const* d_in, const int* in_sizes, int n_in,
                              void* d_out, int out_size, void* d_ws, size_t ws_size,
                              hipStream_t stream) {
  (void)in_sizes; (void)n_in; (void)out_size; (void)ws_size;
  const float* node_feats = (const float*)d_in[0];
  const float* rot        = (const float*)d_in[1];
  const float* trans      = (const float*)d_in[2];
  const float* pair_feats = (const float*)d_in[3];
  const int*   mask       = (const int*)d_in[4];
  const float* ln_s_g = (const float*)d_in[5];
  const float* ln_s_b = (const float*)d_in[6];
  const float* ln_p_g = (const float*)d_in[7];
  const float* ln_p_b = (const float*)d_in[8];
  const float* W_pb   = (const float*)d_in[9];
  const float* W_qkv  = (const float*)d_in[10];
  const float* pw     = (const float*)d_in[11];
  const float* W_out  = (const float*)d_in[12];
  const float* b_out  = (const float*)d_in[13];
  float* out = (float*)d_out;

  char* base = (char*)d_ws;
  size_t off = 0;
  auto alloc = [&](size_t bytes) { void* r = base + off; off = align256(off + bytes); return r; };

  bf16_t* nf_b   = (bf16_t*)alloc((size_t)N_ * NODE_ * 2);
  bf16_t* wq_b   = (bf16_t*)alloc((size_t)QKVO_ * NODE_ * 2);
  bf16_t* wpb_b  = (bf16_t*)alloc((size_t)16 * PAIR_ * 2);
  bf16_t* wout_b = (bf16_t*)alloc((size_t)NODE_ * OUTW_ * 2);
  float*  qkv    = (float*) alloc((size_t)N_ * QKVO_ * 4);
  bf16_t* qk_a   = (bf16_t*)alloc((size_t)H_ * N_ * 32 * 2);
  bf16_t* qk_b   = (bf16_t*)alloc((size_t)H_ * N_ * 32 * 2);
  bf16_t* vsvpT  = (bf16_t*)alloc((size_t)H_ * 48 * N_ * 2);
  float*  qrow   = (float*) alloc((size_t)H_ * N_ * 4);
  float*  kcol   = (float*) alloc((size_t)H_ * N_ * 4);
  float*  resptg = (float*) alloc((size_t)N_ * H_ * 24 * 4);
  bf16_t* outb   = (bf16_t*)alloc((size_t)N_ * OUTW_ * 2);
  float*  logits = (float*) alloc((size_t)H_ * N_ * N_ * 4);  // bias -> logits in place
  bf16_t* attn   = (bf16_t*)alloc((size_t)H_ * N_ * N_ * 2);

  k_ln_node<<<N_, 128, 0, stream>>>(node_feats, ln_s_g, ln_s_b, nf_b);
  k_f2b<<<(QKVO_ * NODE_ + 255) / 256, 256, 0, stream>>>(W_qkv, wq_b, QKVO_ * NODE_);
  k_f2b<<<(NODE_ * OUTW_ + 255) / 256, 256, 0, stream>>>(W_out, wout_b, NODE_ * OUTW_);
  k_wpb<<<8, 256, 0, stream>>>(W_pb, wpb_b);
  k_gemm_qkv<<<1152, 128, 0, stream>>>(nf_b, wq_b, qkv);
  k_points<<<N_, 384, 0, stream>>>(qkv, rot, trans, pw, qk_a, qk_b, vsvpT, qrow, kcol);
  k_pairbias<<<dim3(N_, 32), 64, 0, stream>>>(pair_feats, ln_p_g, ln_p_b, wpb_b, logits);
  k_logits<<<dim3(16, 64, H_), 32, 0, stream>>>(qk_a, qk_b, qrow, kcol, mask, logits);
  k_softmax<<<dim3(N_, H_), 256, 0, stream>>>(logits, attn);
  k_av<<<dim3(64, H_), 96, 0, stream>>>(attn, vsvpT, outb, resptg);
  k_inv<<<N_, 96, 0, stream>>>(resptg, rot, trans, outb);
  k_respair<<<N_, 256, 0, stream>>>(pair_feats, ln_p_g, ln_p_b, attn, outb);
  k_outgemm<<<384, 128, 0, stream>>>(outb, wout_b, b_out, out);
}
